// SchNet_52991306498535
// MI455X (gfx1250) — compile-verified, matrix operand order checked by reading, not run on hardware
//
#include <hip/hip_runtime.h>
#include <hip/hip_bf16.h>

#define N_NODES 20000
#define E_EDGES 640000
#define G_GRAPHS 16
#define NAF 21
#define H 128
#define NG 50
#define L_LAYERS 6
#define NC 97
#define LDH 136   // padded f16 leading dim for row-major data tiles (272B rows, 16B aligned)
#define LDX 40    // padded leading dim for K=32 embed tiles (80B rows, 16B aligned)

typedef __attribute__((ext_vector_type(16))) _Float16 v16h;
typedef __attribute__((ext_vector_type(8)))  _Float16 v8h;
typedef __attribute__((ext_vector_type(4)))  _Float16 v4h;
typedef __attribute__((ext_vector_type(8)))  float    v8f;
typedef __attribute__((ext_vector_type(4)))  float    v4f;
typedef __attribute__((ext_vector_type(4)))  int      v4i;

#if defined(__has_builtin)
#  if __has_builtin(__builtin_amdgcn_global_load_async_to_lds_b128)
#    define HAVE_ASYNC_CP 1
#  endif
#endif
#ifndef HAVE_ASYNC_CP
#  define HAVE_ASYNC_CP 0
#endif

__device__ __forceinline__ void cp16_g2l(_Float16* l, const _Float16* g) {
#if HAVE_ASYNC_CP
  typedef __attribute__((address_space(1))) v4i gv4i;
  typedef __attribute__((address_space(3))) v4i lv4i;
  __builtin_amdgcn_global_load_async_to_lds_b128((gv4i*)(void*)g, (lv4i*)(void*)l,
                                                 0, 0);
#else
  *(v8h*)l = *(const v8h*)g;
#endif
}

__device__ __forceinline__ void wait_async() {
#if HAVE_ASYNC_CP
#  if defined(__has_builtin) && __has_builtin(__builtin_amdgcn_s_wait_asynccnt)
  __builtin_amdgcn_s_wait_asynccnt(0);
#  else
  asm volatile("s_wait_asynccnt 0x0" ::: "memory");
#  endif
#endif
}

// Copy nHalf f16 (multiple of 1024) from packed global weights into LDS.
__device__ __forceinline__ void stage_lds_copy(_Float16* dst, const _Float16* src,
                                               int nHalf, int tid) {
  for (int i = tid * 8; i < nHalf; i += 128 * 8) cp16_g2l(dst + i, src + i);
}

__device__ __forceinline__ float sspf(float x) {
  return fmaxf(x, 0.0f) + log1pf(__expf(-fabsf(x))) - 0.69314718056f;
}

__device__ __forceinline__ v8f wmma16(v16h a, v16h b, v8f c) {
  return __builtin_amdgcn_wmma_f32_16x16x32_f16(false, a, false, b, (short)0, c,
                                                false, false);
}

__device__ __forceinline__ v16h cat16(v8h lo, v8h hi) {
  return __builtin_shufflevector(lo, hi, 0, 1, 2, 3, 4, 5, 6, 7,
                                 8, 9, 10, 11, 12, 13, 14, 15);
}

// Fragment from row-major tile: 8 halfs at p, 8 halfs at p+32B.
__device__ __forceinline__ v16h frag_row(const _Float16* p) {
  return cat16(*(const v8h*)p, *(const v8h*)(p + 16));
}

// Fragment from packed layout: 32 contiguous bytes per lane.
__device__ __forceinline__ v16h frag_pk(const _Float16* p) {
  return cat16(*(const v8h*)p, *(const v8h*)(p + 8));
}

// Bias fragment: 8 consecutive output features (rows of D tile).
__device__ __forceinline__ v8f bias_frag(const float* b, int off) {
  v4f x = *(const v4f*)(b + off);
  v4f y = *(const v4f*)(b + off + 4);
  v8f r = {x[0], x[1], x[2], x[3], y[0], y[1], y[2], y[3]};
  return r;
}

// Stage 64 data rows (global f32) into row-major f16 LDS tile with pad.
__device__ __forceinline__ void stage_rows64(_Float16* dst, const float* __restrict__ A,
                                             int rb, int tid) {
  for (int i = tid; i < 64 * 32; i += 128) {
    int r = i >> 5, kq = i & 31;
    int gr = rb + r;
    v4h v = {};
    if (gr < N_NODES) {
      v4f x = *(const v4f*)&A[gr * H + kq * 4];
      v[0] = (_Float16)x[0]; v[1] = (_Float16)x[1];
      v[2] = (_Float16)x[2]; v[3] = (_Float16)x[3];
    }
    *(v4h*)&dst[r * LDH + kq * 4] = v;
  }
}

// Store D fragment (8 consecutive K-features of one column) into packed-B layout.
__device__ __forceinline__ void store_pk(_Float16* base, int jt, int lane, v8f d) {
  v8h h;
#pragma unroll
  for (int i = 0; i < 8; ++i) h[i] = (_Float16)d[i];
  *(v8h*)(base + (((jt >> 1) * 32 + lane) * 16) + ((jt & 1) << 3)) = h;
}

// ------------------------- pack W[kTot x H] (f32, kMax valid rows) into
// A-fragment-packed f16 global buffer: tile (jt,kc) -> 32 lanes x 16 halfs.
__global__ void pack_w_kernel(const float* __restrict__ W, _Float16* __restrict__ dst,
                              int kMax, int kTot) {
  int nkc = kTot >> 5;
  int i = blockIdx.x * blockDim.x + threadIdx.x;
  if (i >= (kTot >> 2) * H) return;
  int kq = i >> 7, j = i & 127;
  int k0 = kq << 2;
  int jt = j >> 4, r = j & 15;
  int kc = k0 >> 5, kl = k0 & 31;
  int lo = (kl >> 3) & 1;
  int pos = (kl & 7) | ((kl >> 4) << 3);
  int lane = lo * 16 + r;
  v4h v;
#pragma unroll
  for (int t = 0; t < 4; ++t)
    v[t] = (k0 + t < kMax) ? (_Float16)W[(k0 + t) * H + j] : (_Float16)0.0f;
  *(v4h*)&dst[(((jt * nkc + kc) * 32) + lane) * 16 + pos] = v;
}

// ---------------------------------------------------------------- edge geometry
__global__ void edge_geom_kernel(const float* __restrict__ pos,
                                 const long long* __restrict__ ei,
                                 float* __restrict__ ew, float* __restrict__ cc) {
  int e = blockIdx.x * blockDim.x + threadIdx.x;
  if (e >= E_EDGES) return;
  int r = (int)ei[e], c = (int)ei[E_EDGES + e];
  float dx = pos[r * 3 + 0] - pos[c * 3 + 0];
  float dy = pos[r * 3 + 1] - pos[c * 3 + 1];
  float dz = pos[r * 3 + 2] - pos[c * 3 + 2];
  float w = sqrtf(dx * dx + dy * dy + dz * dz);
  ew[e] = w;
  cc[e] = 0.5f * (__cosf(w * (3.14159265358979f / 10.0f)) + 1.0f);
}

__global__ void zero_kernel(float* __restrict__ p, int n) {
  int i = blockIdx.x * blockDim.x + threadIdx.x;
  if (i < n) p[i] = 0.0f;
}

// ------------------------------------------------- h = x @ emb_w + b  (K=21->32)
__global__ void __launch_bounds__(128)
embed_kernel(const float* __restrict__ x, const _Float16* __restrict__ wpkg,
             const float* __restrict__ b, float* __restrict__ Out) {
  __shared__ __align__(16) _Float16 rows[64 * LDX];
  __shared__ __align__(16) _Float16 wpk[32 * H];
  int tid = threadIdx.x, lane = tid & 31, wv = tid >> 5;
  int c0 = lane & 15, r0 = (lane >> 4) << 3, kb = r0;
  int rb = blockIdx.x * 64;
  for (int i = tid; i < 64 * 32; i += 128) {
    int r = i >> 5, k = i & 31;
    int gr = rb + r;
    float v = (gr < N_NODES && k < NAF) ? x[gr * NAF + k] : 0.0f;
    rows[r * LDX + k] = (_Float16)v;
  }
  stage_lds_copy(wpk, wpkg, 32 * H, tid);
  wait_async();
  __syncthreads();
  v8f acc[8];
#pragma unroll
  for (int jt = 0; jt < 8; ++jt) acc[jt] = bias_frag(b, jt * 16 + r0);
  v16h bf = frag_row(&rows[(wv * 16 + c0) * LDX + kb]);
#pragma unroll
  for (int jt = 0; jt < 8; ++jt) {
    v16h a = frag_pk(&wpk[(jt * 32 + lane) * 16]);
    acc[jt] = wmma16(a, bf, acc[jt]);
  }
  int gr = rb + wv * 16 + c0;
  if (gr < N_NODES) {
#pragma unroll
    for (int jt = 0; jt < 8; ++jt) {
      int off = jt * 16 + r0;
      v4f p0 = {acc[jt][0], acc[jt][1], acc[jt][2], acc[jt][3]};
      v4f p1 = {acc[jt][4], acc[jt][5], acc[jt][6], acc[jt][7]};
      *(v4f*)&Out[gr * H + off] = p0;
      *(v4f*)&Out[gr * H + off + 4] = p1;
    }
  }
}

// ---------------------------------------------------------------- hx = h @ c1w
__global__ void __launch_bounds__(128)
node_gemm_kernel(const float* __restrict__ A, const _Float16* __restrict__ wpkg,
                 float* __restrict__ Out) {
  __shared__ __align__(16) _Float16 rows[64 * LDH];
  __shared__ __align__(16) _Float16 wpk[H * H];
  int tid = threadIdx.x, lane = tid & 31, wv = tid >> 5;
  int c0 = lane & 15, r0 = (lane >> 4) << 3, kb = r0;
  int rb = blockIdx.x * 64;
  stage_rows64(rows, A, rb, tid);
  stage_lds_copy(wpk, wpkg, H * H, tid);
  wait_async();
  __syncthreads();
  v8f acc[8] = {};
#pragma unroll
  for (int kc = 0; kc < 4; ++kc) {
    v16h b = frag_row(&rows[(wv * 16 + c0) * LDH + kc * 32 + kb]);
#pragma unroll
    for (int jt = 0; jt < 8; ++jt) {
      v16h a = frag_pk(&wpk[((jt * 4 + kc) * 32 + lane) * 16]);
      acc[jt] = wmma16(a, b, acc[jt]);
    }
  }
  int gr = rb + wv * 16 + c0;
  if (gr < N_NODES) {
#pragma unroll
    for (int jt = 0; jt < 8; ++jt) {
      int off = jt * 16 + r0;
      v4f p0 = {acc[jt][0], acc[jt][1], acc[jt][2], acc[jt][3]};
      v4f p1 = {acc[jt][4], acc[jt][5], acc[jt][6], acc[jt][7]};
      *(v4f*)&Out[gr * H + off] = p0;
      *(v4f*)&Out[gr * H + off + 4] = p1;
    }
  }
}

// ---------------------------------------------------------------- fused edge filter
__global__ void __launch_bounds__(128)
edge_filter_kernel(const float* __restrict__ ew, const float* __restrict__ cc,
                   const long long* __restrict__ ei,
                   const _Float16* __restrict__ w1pkg, const float* __restrict__ m1b,
                   const _Float16* __restrict__ w2pkg, const float* __restrict__ m2b,
                   const float* __restrict__ hx, float* __restrict__ agg) {
  __shared__ __align__(16) _Float16 w1pk[64 * H];   // reused as packed t after barrier
  __shared__ __align__(16) _Float16 w2pk[H * H];
  int tid = threadIdx.x, lane = tid & 31, wv = tid >> 5;
  int c0 = lane & 15, r0 = (lane >> 4) << 3, kb = r0;

  __builtin_prefetch(hx + tid, 0, 1);
  stage_lds_copy(w1pk, w1pkg, 64 * H, tid);
  stage_lds_copy(w2pk, w2pkg, H * H, tid);
  wait_async();
  __syncthreads();

  int e = (blockIdx.x * 4 + wv) * 16 + c0;   // E % 64 == 0: no tail
  float w  = ew[e];
  float Cm = cc[e];
  int src = (int)ei[e];
  int dst = (int)ei[E_EDGES + e];

  // ---- GEMM1: t = ssp(m1w^T x gauss + b1)
  v8f acc[8];
#pragma unroll
  for (int jt = 0; jt < 8; ++jt) acc[jt] = bias_frag(m1b, jt * 16 + r0);
  const float delta = 10.0f / 49.0f;
  const float coeff = -0.5f / (delta * delta);
#pragma unroll
  for (int kc = 0; kc < 2; ++kc) {
    v16h b;
#pragma unroll
    for (int i = 0; i < 8; ++i) {
      int k0 = kc * 32 + kb + i, k1 = k0 + 16;
      float d0 = w - (float)k0 * delta;
      float d1 = w - (float)k1 * delta;
      b[i]     = (k0 < NG) ? (_Float16)__expf(coeff * d0 * d0) : (_Float16)0.0f;
      b[i + 8] = (k1 < NG) ? (_Float16)__expf(coeff * d1 * d1) : (_Float16)0.0f;
    }
#pragma unroll
    for (int jt = 0; jt < 8; ++jt) {
      v16h a = frag_pk(&w1pk[((jt * 2 + kc) * 32 + lane) * 16]);
      acc[jt] = wmma16(a, b, acc[jt]);
    }
  }
#pragma unroll
  for (int jt = 0; jt < 8; ++jt)
#pragma unroll
    for (int i = 0; i < 8; ++i) acc[jt][i] = sspf(acc[jt][i]);

  __syncthreads();            // all waves done reading w1pk -> reuse as packed t
  _Float16* tw = &w1pk[wv * 2048];
#pragma unroll
  for (int jt = 0; jt < 8; ++jt) store_pk(tw, jt, lane, acc[jt]);

  // ---- GEMM2: Wf = (m2w^T x t + b2) * C
  v8f acc2[8];
#pragma unroll
  for (int jt = 0; jt < 8; ++jt) acc2[jt] = bias_frag(m2b, jt * 16 + r0);
#pragma unroll
  for (int kc = 0; kc < 4; ++kc) {
    v16h b = frag_pk(tw + (kc * 32 + lane) * 16);
#pragma unroll
    for (int jt = 0; jt < 8; ++jt) {
      v16h a = frag_pk(&w2pk[((jt * 4 + kc) * 32 + lane) * 16]);
      acc2[jt] = wmma16(a, b, acc2[jt]);
    }
  }

  // ---- gather hx[src], modulate, scatter into agg[dst]
  const float* hr = hx + src * H;
  float* ar = agg + dst * H;
#pragma unroll
  for (int jt = 0; jt < 8; ++jt) {
    int off = jt * 16 + r0;
    v4f h0 = *(const v4f*)(hr + off);
    v4f h1 = *(const v4f*)(hr + off + 4);
#pragma unroll
    for (int t = 0; t < 4; ++t)
      __hip_atomic_fetch_add(ar + off + t, acc2[jt][t] * Cm * h0[t],
                             __ATOMIC_RELAXED, __HIP_MEMORY_SCOPE_AGENT);
#pragma unroll
    for (int t = 0; t < 4; ++t)
      __hip_atomic_fetch_add(ar + off + 4 + t, acc2[jt][t + 4] * Cm * h1[t],
                             __ATOMIC_RELAXED, __HIP_MEMORY_SCOPE_AGENT);
  }
}

// -------------------------------------- h += (ssp(agg@c2w + c2b)) @ lw + lb
__global__ void __launch_bounds__(128)
node_update_kernel(const float* __restrict__ agg,
                   const _Float16* __restrict__ c2pkg, const float* __restrict__ c2b,
                   const _Float16* __restrict__ lpkg, const float* __restrict__ lb,
                   float* __restrict__ h) {
  __shared__ __align__(16) _Float16 rows[64 * LDH];  // agg rows; reused as packed u
  __shared__ __align__(16) _Float16 wpk[H * H];
  int tid = threadIdx.x, lane = tid & 31, wv = tid >> 5;
  int c0 = lane & 15, r0 = (lane >> 4) << 3, kb = r0;
  int rb = blockIdx.x * 64;
  stage_rows64(rows, agg, rb, tid);
  stage_lds_copy(wpk, c2pkg, H * H, tid);
  wait_async();
  __syncthreads();
  v8f acc[8];
#pragma unroll
  for (int jt = 0; jt < 8; ++jt) acc[jt] = bias_frag(c2b, jt * 16 + r0);
#pragma unroll
  for (int kc = 0; kc < 4; ++kc) {
    v16h b = frag_row(&rows[(wv * 16 + c0) * LDH + kc * 32 + kb]);
#pragma unroll
    for (int jt = 0; jt < 8; ++jt) {
      v16h a = frag_pk(&wpk[((jt * 4 + kc) * 32 + lane) * 16]);
      acc[jt] = wmma16(a, b, acc[jt]);
    }
  }
#pragma unroll
  for (int jt = 0; jt < 8; ++jt)
#pragma unroll
    for (int i = 0; i < 8; ++i) acc[jt][i] = sspf(acc[jt][i]);

  __syncthreads();            // rows + wpk fully consumed
  _Float16* uw = &rows[wv * 2048];
#pragma unroll
  for (int jt = 0; jt < 8; ++jt) store_pk(uw, jt, lane, acc[jt]);
  stage_lds_copy(wpk, lpkg, H * H, tid);
  wait_async();
  __syncthreads();

  v8f acc2[8];
#pragma unroll
  for (int jt = 0; jt < 8; ++jt) acc2[jt] = bias_frag(lb, jt * 16 + r0);
#pragma unroll
  for (int kc = 0; kc < 4; ++kc) {
    v16h b = frag_pk(uw + (kc * 32 + lane) * 16);
#pragma unroll
    for (int jt = 0; jt < 8; ++jt) {
      v16h a = frag_pk(&wpk[((jt * 4 + kc) * 32 + lane) * 16]);
      acc2[jt] = wmma16(a, b, acc2[jt]);
    }
  }
  int gr = rb + wv * 16 + c0;
  if (gr < N_NODES) {
#pragma unroll
    for (int jt = 0; jt < 8; ++jt) {
      int off = jt * 16 + r0;
      v4f p0 = *(v4f*)&h[gr * H + off];
      v4f p1 = *(v4f*)&h[gr * H + off + 4];
#pragma unroll
      for (int t = 0; t < 4; ++t) { p0[t] += acc2[jt][t]; p1[t] += acc2[jt][t + 4]; }
      *(v4f*)&h[gr * H + off] = p0;
      *(v4f*)&h[gr * H + off + 4] = p1;
    }
  }
}

// ---------------------------- relu(h@lin1+b1) pooled by batch via L2 atomics
__global__ void __launch_bounds__(128)
head1_kernel(const float* __restrict__ h, const _Float16* __restrict__ w1pkg,
             const float* __restrict__ b1, const long long* __restrict__ batch,
             float* __restrict__ pool) {
  __shared__ __align__(16) _Float16 rows[64 * LDH];
  __shared__ __align__(16) _Float16 wpk[H * H];
  int tid = threadIdx.x, lane = tid & 31, wv = tid >> 5;
  int c0 = lane & 15, r0 = (lane >> 4) << 3, kb = r0;
  int rb = blockIdx.x * 64;
  stage_rows64(rows, h, rb, tid);
  stage_lds_copy(wpk, w1pkg, H * H, tid);
  wait_async();
  __syncthreads();
  v8f acc[8];
#pragma unroll
  for (int jt = 0; jt < 8; ++jt) acc[jt] = bias_frag(b1, jt * 16 + r0);
#pragma unroll
  for (int kc = 0; kc < 4; ++kc) {
    v16h b = frag_row(&rows[(wv * 16 + c0) * LDH + kc * 32 + kb]);
#pragma unroll
    for (int jt = 0; jt < 8; ++jt) {
      v16h a = frag_pk(&wpk[((jt * 4 + kc) * 32 + lane) * 16]);
      acc[jt] = wmma16(a, b, acc[jt]);
    }
  }
  int gr = rb + wv * 16 + c0;
  if (gr < N_NODES) {
    int g = (int)batch[gr];
    float* pp = pool + g * H;
#pragma unroll
    for (int jt = 0; jt < 8; ++jt) {
      int off = jt * 16 + r0;
#pragma unroll
      for (int i = 0; i < 8; ++i)
        __hip_atomic_fetch_add(pp + off + i, fmaxf(acc[jt][i], 0.0f),
                               __ATOMIC_RELAXED, __HIP_MEMORY_SCOPE_AGENT);
    }
  }
}

__global__ void count_kernel(const long long* __restrict__ batch,
                             float* __restrict__ cnt) {
  int n = blockIdx.x * blockDim.x + threadIdx.x;
  if (n < N_NODES)
    __hip_atomic_fetch_add(&cnt[(int)batch[n]], 1.0f,
                           __ATOMIC_RELAXED, __HIP_MEMORY_SCOPE_AGENT);
}

// ----------------------------- tiny head: mean, relu(ge@lin1+b), @lin2+b
__global__ void __launch_bounds__(128)
head2_kernel(const float* __restrict__ pool, const float* __restrict__ cnt,
             const float* __restrict__ w1, const float* __restrict__ b1,
             const float* __restrict__ w2, const float* __restrict__ b2,
             float* __restrict__ out) {
  __shared__ float ge[G_GRAPHS][H];
  __shared__ float h2[G_GRAPHS][H];
  int j = threadIdx.x;
  for (int g = 0; g < G_GRAPHS; ++g) {
    float v = pool[g * H + j] / fmaxf(cnt[g], 1.0f);
    ge[g][j] = v;
    out[g * H + j] = v;   // graph_emb output
  }
  __syncthreads();
  for (int g = 0; g < G_GRAPHS; ++g) {
    float s = b1[j];
#pragma unroll 8
    for (int k = 0; k < H; ++k) s = fmaf(ge[g][k], w1[k * H + j], s);
    h2[g][j] = fmaxf(s, 0.0f);
  }
  __syncthreads();
  float* out2 = out + G_GRAPHS * H;
  if (j < NC) {
    for (int g = 0; g < G_GRAPHS; ++g) {
      float s = b2[j];
#pragma unroll 8
      for (int k = 0; k < H; ++k) s = fmaf(h2[g][k], w2[k * NC + j], s);
      out2[g * NC + j] = s;
    }
  }
}

extern "C" void kernel_launch(void* const* d_in, const int* in_sizes, int n_in,
                              void* d_out, int out_size, void* d_ws, size_t ws_size,
                              hipStream_t stream) {
  (void)in_sizes; (void)n_in; (void)out_size; (void)ws_size;
  const float*     x      = (const float*)d_in[0];
  const float*     pos    = (const float*)d_in[1];
  const long long* ei     = (const long long*)d_in[2];
  const long long* batch  = (const long long*)d_in[3];
  const float* emb_w  = (const float*)d_in[4];
  const float* emb_b  = (const float*)d_in[5];
  const float* mlp1_w = (const float*)d_in[6];
  const float* mlp1_b = (const float*)d_in[7];
  const float* mlp2_w = (const float*)d_in[8];
  const float* mlp2_b = (const float*)d_in[9];
  const float* conv1_w = (const float*)d_in[10];
  const float* conv2_w = (const float*)d_in[11];
  const float* conv2_b = (const float*)d_in[12];
  const float* lin_w  = (const float*)d_in[13];
  const float* lin_b  = (const float*)d_in[14];
  const float* lin1_w = (const float*)d_in[15];
  const float* lin1_b = (const float*)d_in[16];
  const float* lin2_w = (const float*)d_in[17];
  const float* lin2_b = (const float*)d_in[18];
  float* out = (float*)d_out;

  float* ws   = (float*)d_ws;
  float* h    = ws;                    // N*H
  float* hx   = h + N_NODES * H;       // N*H
  float* agg  = hx + N_NODES * H;      // N*H
  float* ew   = agg + N_NODES * H;     // E
  float* cc   = ew + E_EDGES;          // E
  float* pool = cc + E_EDGES;          // G*H
  float* cnt  = pool + G_GRAPHS * H;   // G (contiguous with pool)

  // packed f16 weight buffers (32B aligned) after the float workspace
  uintptr_t pb = (uintptr_t)(cnt + G_GRAPHS);
  pb = (pb + 31) & ~(uintptr_t)31;
  _Float16* embpk = (_Float16*)pb;                    // 32*H
  _Float16* c1pk  = embpk + 32 * H;                   // L*H*H
  _Float16* w1pk  = c1pk + L_LAYERS * H * H;          // L*64*H
  _Float16* w2pk  = w1pk + L_LAYERS * 64 * H;         // L*H*H
  _Float16* c2pk  = w2pk + L_LAYERS * H * H;          // L*H*H
  _Float16* lpk   = c2pk + L_LAYERS * H * H;          // L*H*H
  _Float16* l1pk  = lpk + L_LAYERS * H * H;           // H*H

  // ---- pack all weights into WMMA-fragment layout (f16) once per call
  pack_w_kernel<<<8, 128, 0, stream>>>(emb_w, embpk, NAF, 32);
  for (int l = 0; l < L_LAYERS; ++l) {
    pack_w_kernel<<<32, 128, 0, stream>>>(conv1_w + l * H * H, c1pk + l * H * H, H, H);
    pack_w_kernel<<<16, 128, 0, stream>>>(mlp1_w + l * NG * H, w1pk + l * 64 * H, NG, 64);
    pack_w_kernel<<<32, 128, 0, stream>>>(mlp2_w + l * H * H, w2pk + l * H * H, H, H);
    pack_w_kernel<<<32, 128, 0, stream>>>(conv2_w + l * H * H, c2pk + l * H * H, H, H);
    pack_w_kernel<<<32, 128, 0, stream>>>(lin_w + l * H * H, lpk + l * H * H, H, H);
  }
  pack_w_kernel<<<32, 128, 0, stream>>>(lin1_w, l1pk, H, H);

  edge_geom_kernel<<<(E_EDGES + 255) / 256, 256, 0, stream>>>(pos, ei, ew, cc);
  embed_kernel<<<(N_NODES + 63) / 64, 128, 0, stream>>>(x, embpk, emb_b, h);

  const int gemm_blocks = (N_NODES + 63) / 64;   // 313
  for (int l = 0; l < L_LAYERS; ++l) {
    zero_kernel<<<(N_NODES * H + 255) / 256, 256, 0, stream>>>(agg, N_NODES * H);
    node_gemm_kernel<<<gemm_blocks, 128, 0, stream>>>(h, c1pk + l * H * H, hx);
    edge_filter_kernel<<<E_EDGES / 64, 128, 0, stream>>>(
        ew, cc, ei, w1pk + l * 64 * H, mlp1_b + l * H,
        w2pk + l * H * H, mlp2_b + l * H, hx, agg);
    node_update_kernel<<<gemm_blocks, 128, 0, stream>>>(
        agg, c2pk + l * H * H, conv2_b + l * H,
        lpk + l * H * H, lin_b + l * H, h);
  }

  int pz = G_GRAPHS * H + G_GRAPHS;
  zero_kernel<<<(pz + 255) / 256, 256, 0, stream>>>(pool, pz);
  count_kernel<<<(N_NODES + 255) / 256, 256, 0, stream>>>(batch, cnt);
  head1_kernel<<<gemm_blocks, 128, 0, stream>>>(h, l1pk, lin1_b, batch, pool);
  head2_kernel<<<1, 128, 0, stream>>>(pool, cnt, lin1_w, lin1_b, lin2_w, lin2_b, out);
}
